// MoELayer_86294482911895
// MI455X (gfx1250) — matrix operand order repaired and checked by reading, not compiled, not run on hardware
//
#include <hip/hip_runtime.h>
#include <hip/hip_bf16.h>
#include <math.h>

// ---------------------------------------------------------------------------
// MoE (Switch top-1) for gfx1250:
//   1) router (logits/softmax/top-1, expert lists, f16 activations)
//   2) weight swizzle: f32 -> f16 in WMMA B-fragment order (1KB per 32x16 tile)
//   3) grouped expert FFN: WMMA f16 16x16x32, f32 accumulate, async LDS staging
// ---------------------------------------------------------------------------

#define NUM_EXPERTS 8
#define D_MODEL     512
#define SEQ         2048
#define D_FF        2048

#define M_TILE      32              // tokens per FFN block
#define XLD         (D_MODEL + 16)  // padded LDS row stride (halfs)
#define HLD         (D_FF + 16)     // padded LDS row stride (halfs)

#define KT1 (D_MODEL / 32)          // 16  K-tiles, phase 1
#define NT1 (D_FF / 16)             // 128 N-tiles, phase 1
#define KT2 (D_FF / 32)             // 64  K-tiles, phase 2
#define NT2 (D_MODEL / 16)          // 32  N-tiles, phase 2
#define TILE_HALFS 512              // 32 lanes * 16 halfs per (kt,nt) B tile

typedef _Float16 v16h __attribute__((ext_vector_type(16)));
typedef _Float16 v8h  __attribute__((ext_vector_type(8)));
typedef float    v8f  __attribute__((ext_vector_type(8)));

// ---------------------------------------------------------------------------
// A-fragment (16x32 f16) from LDS: lane m / m+16 hold row m.
// lanes 0-15: K = 0..15 ; lanes 16-31: K = 16..31.
// ---------------------------------------------------------------------------
__device__ __forceinline__ v16h load_A_lds(const _Float16* __restrict__ arow,
                                           int kbase, int hi) {
  const v8h lo = *(const v8h*)(arow + kbase + hi * 16);
  const v8h up = *(const v8h*)(arow + kbase + 8 + hi * 16);
  v16h a;
#pragma unroll
  for (int i = 0; i < 8; ++i) { a[i] = lo[i]; a[i + 8] = up[i]; }
  return a;
}

// ---------------------------------------------------------------------------
// Kernel 1: router
// ---------------------------------------------------------------------------
__global__ __launch_bounds__(256)
void moe_router_kernel(const float* __restrict__ H,      // [SEQ, D_MODEL]
                       const float* __restrict__ Wr,     // [D_MODEL, NUM_EXPERTS]
                       float* __restrict__ logits_out,   // [SEQ, NUM_EXPERTS]
                       int*   __restrict__ expert_out,   // [SEQ]
                       float* __restrict__ probs_ws,     // [SEQ]
                       int*   __restrict__ counts,       // [NUM_EXPERTS]
                       int*   __restrict__ lists,        // [NUM_EXPERTS, SEQ]
                       _Float16* __restrict__ Xh)        // [SEQ, D_MODEL] f16
{
  __shared__ float wr[D_MODEL * NUM_EXPERTS];
  for (int i = threadIdx.x; i < D_MODEL * NUM_EXPERTS; i += blockDim.x)
    wr[i] = Wr[i];
  __syncthreads();

  const int t = blockIdx.x * blockDim.x + threadIdx.x;
  if (t >= SEQ) return;

  const float* __restrict__ x = H + (size_t)t * D_MODEL;
  float acc[NUM_EXPERTS];
#pragma unroll
  for (int e = 0; e < NUM_EXPERTS; ++e) acc[e] = 0.0f;

  for (int d = 0; d < D_MODEL; ++d) {
    const float xv = x[d];
    Xh[(size_t)t * D_MODEL + d] = (_Float16)xv;
#pragma unroll
    for (int e = 0; e < NUM_EXPERTS; ++e)
      acc[e] = fmaf(xv, wr[d * NUM_EXPERTS + e], acc[e]);
  }

  float m = acc[0];
  int best = 0;
#pragma unroll
  for (int e = 1; e < NUM_EXPERTS; ++e)
    if (acc[e] > m) { m = acc[e]; best = e; }
  float s = 0.0f;
#pragma unroll
  for (int e = 0; e < NUM_EXPERTS; ++e) s += expf(acc[e] - m);
  const float prob = 1.0f / s;   // exp(m-m)/sum

#pragma unroll
  for (int e = 0; e < NUM_EXPERTS; ++e)
    logits_out[(size_t)t * NUM_EXPERTS + e] = acc[e];
  expert_out[t] = best;
  probs_ws[t]   = prob;

  const int pos = atomicAdd(&counts[best], 1);
  lists[best * SEQ + pos] = t;
}

// ---------------------------------------------------------------------------
// Kernel 2: swizzle f32 weights [E, K, N] into f16 WMMA B-fragment order.
// Layout: block(e, nt, kt) of 512 halfs; within block, lane*16 + j, where
//   element (lane, j) = W[e][kt*32 + ((lane&16)?16:0) + j][nt*16 + (lane&15)].
// One thread per (e, nt, kt, lane) slot -> 16 strided f32 reads, one 32B store.
// ---------------------------------------------------------------------------
__global__ __launch_bounds__(256)
void moe_swizzle_kernel(const float* __restrict__ W, _Float16* __restrict__ Wsw,
                        int K, int N) {
  const int KT = K >> 5, NT = N >> 4;
  const size_t id   = (size_t)blockIdx.x * blockDim.x + threadIdx.x;
  const int    lane = (int)(id & 31);
  const size_t slot = id >> 5;                   // ((e*NT)+nt)*KT + kt
  const int    kt   = (int)(slot % KT);
  const size_t t2   = slot / KT;
  const int    nt   = (int)(t2 % NT);
  const int    e    = (int)(t2 / NT);
  if (e >= NUM_EXPERTS) return;

  const float* __restrict__ p =
      W + (size_t)e * K * N +
      (size_t)(kt * 32 + ((lane & 16) ? 16 : 0)) * N + nt * 16 + (lane & 15);
  v16h b;
#pragma unroll
  for (int j = 0; j < 16; ++j) b[j] = (_Float16)p[(size_t)j * N];
  *(v16h*)(Wsw + (slot * 32 + lane) * 16) = b;
}

// ---------------------------------------------------------------------------
// Kernel 3: grouped expert FFN, 32 tokens/block, swizzled f16 weights.
// ---------------------------------------------------------------------------
__global__ __launch_bounds__(256)
void moe_ffn_kernel(const _Float16* __restrict__ Xh,    // [SEQ, D_MODEL] f16
                    const _Float16* __restrict__ WiSw,  // swizzled [E]
                    const _Float16* __restrict__ WoSw,  // swizzled [E]
                    const int*   __restrict__ counts,
                    const int*   __restrict__ lists,
                    const float* __restrict__ probs,
                    float* __restrict__ out)            // [SEQ, D_MODEL]
{
  extern __shared__ char smem[];
  _Float16* __restrict__ Xl = (_Float16*)smem;           // [M_TILE, XLD]
  _Float16* __restrict__ Hl = Xl + M_TILE * XLD;         // [M_TILE, HLD]

  __shared__ int   tokS[M_TILE];
  __shared__ float probS[M_TILE];
  __shared__ int   validS[M_TILE];

  const int e    = blockIdx.y;
  const int cnt  = counts[e];
  const int base = blockIdx.x * M_TILE;
  if (base >= cnt) return;

  if (threadIdx.x < M_TILE) {
    const int idx = base + (int)threadIdx.x;
    const int v   = (idx < cnt) ? 1 : 0;
    const int tk  = v ? lists[e * SEQ + idx] : 0;
    tokS[threadIdx.x]   = tk;
    validS[threadIdx.x] = v;
    probS[threadIdx.x]  = probs[tk];
  }
  __syncthreads();

  // Stage X tile (M_TILE x D_MODEL halfs) into LDS via async copy (ASYNCcnt).
  {
    const int chunks = M_TILE * (D_MODEL / 8);   // 16B chunks
    for (int c = threadIdx.x; c < chunks; c += blockDim.x) {
      const int row = c >> 6;                    // 64 chunks per row
      const int c8  = c & 63;
      const unsigned long long gaddr =
          (unsigned long long)(uintptr_t)(Xh + (size_t)tokS[row] * D_MODEL + c8 * 8);
      const unsigned ldsaddr =
          (unsigned)(uintptr_t)(Xl + row * XLD + c8 * 8);   // flat low 32 = LDS addr
      asm volatile("global_load_async_to_lds_b128 %0, %1, off"
                   :: "v"(ldsaddr), "v"(gaddr) : "memory");
    }
    asm volatile("s_wait_asynccnt 0x0" ::: "memory");
  }
  __syncthreads();

  const _Float16* __restrict__ WiE = WiSw + (size_t)e * KT1 * NT1 * TILE_HALFS;
  const _Float16* __restrict__ WoE = WoSw + (size_t)e * KT2 * NT2 * TILE_HALFS;

  const int lane = threadIdx.x & 31;
  const int wave = threadIdx.x >> 5;
  const int m    = lane & 15;
  const int hi   = lane >> 4;
  const int col  = lane & 15;
  const int rb   = hi * 8;

  // ---- Phase 1: Hl = relu(Xl @ Wi)   [M_TILE x D_FF] -------------------
  for (int job = wave; job < 2 * NT1; job += 8) {
    const int r0 = (job & 1) * 16;
    const int nt = job >> 1;
    const _Float16* arow = Xl + (r0 + m) * XLD;
    const _Float16* bblk = WiE + (size_t)nt * KT1 * TILE_HALFS + lane * 16;

    v8f c = {};
#pragma unroll 4
    for (int kt = 0; kt < KT1; ++kt) {
      const v16h a = load_A_lds(arow, kt * 32, hi);
      const v16h b = *(const v16h*)(bblk + (size_t)kt * TILE_HALFS);
      // Unconditional speculative prefetch: OOB translations are silently
      // dropped (ISA §10.5), so no guard branch needed in the hot loop.
      __builtin_prefetch(bblk + (size_t)(kt + 4) * TILE_HALFS, 0, 1);
      c = __builtin_amdgcn_wmma_f32_16x16x32_f16(false, a, false, b,
                                                 (short)0, c, false, false);
    }
#pragma unroll
    for (int r = 0; r < 8; ++r)
      Hl[(r0 + rb + r) * HLD + nt * 16 + col] = (_Float16)fmaxf(c[r], 0.0f);
  }
  __syncthreads();

  // ---- Phase 2: out = prob * (Hl @ Wo)   [M_TILE x D_MODEL] ------------
  for (int job = wave; job < 2 * NT2; job += 8) {
    const int r0 = (job & 1) * 16;
    const int nt = job >> 1;
    const _Float16* arow = Hl + (r0 + m) * HLD;
    const _Float16* bblk = WoE + (size_t)nt * KT2 * TILE_HALFS + lane * 16;

    v8f c = {};
#pragma unroll 4
    for (int kt = 0; kt < KT2; ++kt) {
      const v16h a = load_A_lds(arow, kt * 32, hi);
      const v16h b = *(const v16h*)(bblk + (size_t)kt * TILE_HALFS);
      __builtin_prefetch(bblk + (size_t)(kt + 4) * TILE_HALFS, 0, 1);
      c = __builtin_amdgcn_wmma_f32_16x16x32_f16(false, a, false, b,
                                                 (short)0, c, false, false);
    }
#pragma unroll
    for (int r = 0; r < 8; ++r) {
      const int row = r0 + rb + r;
      if (validS[row])
        out[(size_t)tokS[row] * D_MODEL + nt * 16 + col] = probS[row] * c[r];
    }
  }
}

// ---------------------------------------------------------------------------
// Host-side launch
// ---------------------------------------------------------------------------
extern "C" void kernel_launch(void* const* d_in, const int* in_sizes, int n_in,
                              void* d_out, int out_size, void* d_ws, size_t ws_size,
                              hipStream_t stream) {
  (void)in_sizes; (void)n_in; (void)out_size; (void)ws_size;

  const float* H  = (const float*)d_in[0];   // [1, SEQ, D_MODEL]
  const float* Wr = (const float*)d_in[1];   // [D_MODEL, NUM_EXPERTS]
  const float* wi = (const float*)d_in[2];   // [E, D_MODEL, D_FF]
  const float* wo = (const float*)d_in[3];   // [E, D_FF, D_MODEL]

  // Output: out (SEQ*D_MODEL f32) | router_logits (SEQ*8 f32) | expert_index (SEQ i32)
  float* out        = (float*)d_out;
  float* logits_out = out + (size_t)SEQ * D_MODEL;
  int*   expert_out = (int*)(logits_out + (size_t)SEQ * NUM_EXPERTS);

  // Workspace layout (bytes):
  //   [0,        8192)            probs   f32[SEQ]
  //   [8192,     8224)            counts  i32[8]
  //   [8320,     73856)           lists   i32[8][SEQ]
  //   [73856,    2171008)         Xh      f16[SEQ][D_MODEL]
  //   [2171008,  18948224)        WiSw    f16 swizzled [E][K*N]
  //   [18948224, 35725440)        WoSw    f16 swizzled [E][K*N]
  char* ws = (char*)d_ws;
  float*    probs  = (float*)(ws + 0);
  int*      counts = (int*)(ws + 8192);
  int*      lists  = (int*)(ws + 8320);
  _Float16* Xh     = (_Float16*)(ws + 73856);
  _Float16* WiSw   = (_Float16*)(ws + 2171008);
  _Float16* WoSw   = (_Float16*)(ws + 18948224);

  hipMemsetAsync(counts, 0, NUM_EXPERTS * sizeof(int), stream);

  moe_router_kernel<<<SEQ / 256, 256, 0, stream>>>(
      H, Wr, logits_out, expert_out, probs, counts, lists, Xh);

  // Swizzle wi (K=512, N=2048) and wo (K=2048, N=512): 2048 blocks each.
  {
    const int threads_wi = NUM_EXPERTS * KT1 * NT1 * 32;
    const int threads_wo = NUM_EXPERTS * KT2 * NT2 * 32;
    moe_swizzle_kernel<<<threads_wi / 256, 256, 0, stream>>>(wi, WiSw, D_MODEL, D_FF);
    moe_swizzle_kernel<<<threads_wo / 256, 256, 0, stream>>>(wo, WoSw, D_FF, D_MODEL);
  }

  const size_t smem_bytes = (size_t)(M_TILE * XLD + M_TILE * HLD) * sizeof(_Float16);
  dim3 grid(SEQ / M_TILE, NUM_EXPERTS);
  moe_ffn_kernel<<<grid, 256, smem_bytes, stream>>>(
      Xh, WiSw, WoSw, counts, lists, probs, out);
}
